// RTNMarlinLinear_37082747634166
// MI455X (gfx1250) — matrix operand order repaired and checked by reading, not compiled, not run on hardware
//
#include <hip/hip_runtime.h>

// ---------------------------------------------------------------------------
// RTN 4-bit dequant GEMM:  out[M,O] = x[M,K] * ((q - 8) * scale)^T
// M=4096, K=4096, O=11008, group_size=128 (one scale per 128 K per out row)
// bf16 WMMA (v_wmma_f32_16x16x32_bf16), wave32, double-buffered LDS.
// Block tile 256x256 (512 thr = 16 waves), wave tile 64x64 (4x4 16x16 accs).
// ---------------------------------------------------------------------------

typedef __attribute__((ext_vector_type(16))) __bf16 v16bf;
typedef __attribute__((ext_vector_type(8)))  __bf16 v8bf;
typedef __attribute__((ext_vector_type(4)))  __bf16 v4bf;
typedef __attribute__((ext_vector_type(8)))  float  v8f;

#define M_ROWS   4096
#define IN_F     4096
#define OUT_F    11008
#define NGROUPS  32
#define GRP      128

#define BM  256          // block tile M
#define BO  256          // block tile O (N)
#define BK  64           // K per stage (divides GRP -> one scale per stage row)
#define LDK 72           // padded LDS row stride (bf16 elems), 144B = 16B-aligned

__global__ __launch_bounds__(512)
void rtn_marlin_wmma_bf16(const float* __restrict__ x,
                          const int*   __restrict__ qw,
                          const float* __restrict__ scales,
                          float*       __restrict__ out)
{
    __shared__ __attribute__((aligned(16))) __bf16 As[2][BM][LDK];   // 72 KB
    __shared__ __attribute__((aligned(16))) __bf16 Bs[2][BO][LDK];   // 72 KB

    const int tid   = threadIdx.x;
    const int lane  = tid & 31;
    const int wave  = tid >> 5;        // 0..15
    const int l16   = lane & 15;
    const int lhalf = lane >> 4;       // 0 or 1
    const int wm    = wave >> 2;       // 0..3 -> 64 rows of M each
    const int wo    = wave & 3;        // 0..3 -> 64 cols of O each

    const int bo0 = blockIdx.x * BO;
    const int bm0 = blockIdx.y * BM;

    // tile loaders: 16 threads per row (float4/int4) -> 32 rows/pass, 8 passes
    const int lrow = tid >> 4;         // 0..31
    const int lcol = (tid & 15) << 2;  // 0,4,...,60

    v8f acc[4][4];
#pragma unroll
    for (int fm = 0; fm < 4; ++fm)
#pragma unroll
        for (int fn = 0; fn < 4; ++fn)
            acc[fm][fn] = (v8f){0.f, 0.f, 0.f, 0.f, 0.f, 0.f, 0.f, 0.f};

    const int NKT = IN_F / BK;         // 64 stages

    // ---------------- prologue: stage 0 straight into LDS buf 0 -------------
    {
#pragma unroll
        for (int it = 0; it < 8; ++it) {
            const int row = lrow + it * 32;
            const float4 av = *(const float4*)(x  + (size_t)(bm0 + row) * IN_F + lcol);
            const int4   qv = *(const int4*)  (qw + (size_t)(bo0 + row) * IN_F + lcol);
            const float  s  = scales[(bo0 + row) * NGROUPS + 0];
            v4bf a4 = { (__bf16)av.x, (__bf16)av.y, (__bf16)av.z, (__bf16)av.w };
            v4bf b4 = { (__bf16)((float)(qv.x - 8) * s), (__bf16)((float)(qv.y - 8) * s),
                        (__bf16)((float)(qv.z - 8) * s), (__bf16)((float)(qv.w - 8) * s) };
            *(v4bf*)&As[0][row][lcol] = a4;
            *(v4bf*)&Bs[0][row][lcol] = b4;
        }
    }
    __syncthreads();

    // ---------------- main loop: double-buffered ----------------------------
    float4 aReg[8];
    int4   qReg[8];
    float  sReg[8];

    for (int kt = 0; kt < NKT; ++kt) {
        const int  buf      = kt & 1;
        const bool has_next = (kt + 1) < NKT;

        // Phase 1: issue global loads for stage kt+1 into registers
        if (has_next) {
            const int k0 = (kt + 1) * BK;
            const int g  = k0 / GRP;
#pragma unroll
            for (int it = 0; it < 8; ++it) {
                const int row = lrow + it * 32;
                aReg[it] = *(const float4*)(x  + (size_t)(bm0 + row) * IN_F + k0 + lcol);
                qReg[it] = *(const int4*)  (qw + (size_t)(bo0 + row) * IN_F + k0 + lcol);
                sReg[it] = scales[(bo0 + row) * NGROUPS + g];
            }
            if (kt + 2 < NKT) {  // warm L2 for the stage after next
                const int k2 = (kt + 2) * BK;
                __builtin_prefetch(x  + (size_t)(bm0 + lrow) * IN_F + k2 + lcol, 0, 0);
                __builtin_prefetch(qw + (size_t)(bo0 + lrow) * IN_F + k2 + lcol, 0, 0);
            }
        }

        // Phase 2: WMMA over this stage (2 K-steps of 32, 16 WMMAs each)
#pragma unroll
        for (int ks = 0; ks < BK; ks += 32) {
            v16bf afrag[4], bfrag[4];
            // A 16x32 bf16 layout: lanes 0-15 (M=l16) hold K=ks..ks+15,
            //                      lanes 16-31       hold K=ks+16..ks+31
#pragma unroll
            for (int fm = 0; fm < 4; ++fm) {
                const __bf16* base = &As[buf][wm * 64 + fm * 16 + l16][ks + lhalf * 16];
                v8bf lo = *(const v8bf*)(base);
                v8bf hi = *(const v8bf*)(base + 8);
#pragma unroll
                for (int i = 0; i < 8; ++i) { afrag[fm][i] = lo[i]; afrag[fm][i + 8] = hi[i]; }
            }
            // B 32x16 bf16 layout: lanes 0-15 (N=l16) hold K=ks..ks+15,
            //                      lanes 16-31 hold K=ks+16..ks+31 (2 per VGPR)
#pragma unroll
            for (int fn = 0; fn < 4; ++fn) {
                const __bf16* base = &Bs[buf][wo * 64 + fn * 16 + l16][ks + lhalf * 16];
                v8bf lo = *(const v8bf*)(base);
                v8bf hi = *(const v8bf*)(base + 8);
#pragma unroll
                for (int i = 0; i < 8; ++i) { bfrag[fn][i] = lo[i]; bfrag[fn][i + 8] = hi[i]; }
            }
#pragma unroll
            for (int fm = 0; fm < 4; ++fm)
#pragma unroll
                for (int fn = 0; fn < 4; ++fn)
                    acc[fm][fn] = __builtin_amdgcn_wmma_f32_16x16x32_bf16(
                        /*neg_a=*/false, afrag[fm],
                        /*neg_b=*/false, bfrag[fn],
                        /*c_mod=*/(short)0, acc[fm][fn],
                        /*reuse_a=*/false, /*reuse_b=*/false);
        }

        // Phase 3: dequant/convert staged registers into the other LDS buffer
        if (has_next) {
            const int nbuf = buf ^ 1;
#pragma unroll
            for (int it = 0; it < 8; ++it) {
                const int row = lrow + it * 32;
                const float4 av = aReg[it];
                const int4   qv = qReg[it];
                const float  s  = sReg[it];
                v4bf a4 = { (__bf16)av.x, (__bf16)av.y, (__bf16)av.z, (__bf16)av.w };
                v4bf b4 = { (__bf16)((float)(qv.x - 8) * s), (__bf16)((float)(qv.y - 8) * s),
                            (__bf16)((float)(qv.z - 8) * s), (__bf16)((float)(qv.w - 8) * s) };
                *(v4bf*)&As[nbuf][row][lcol] = a4;
                *(v4bf*)&Bs[nbuf][row][lcol] = b4;
            }
        }
        __syncthreads();
    }

    // ---------------- epilogue: C/D layout -> global ------------------------
    // acc VGPR r: lanes 0-15 -> M = r,  lanes 16-31 -> M = r + 8; N = l16
#pragma unroll
    for (int fm = 0; fm < 4; ++fm) {
#pragma unroll
        for (int fn = 0; fn < 4; ++fn) {
            const int o = bo0 + wo * 64 + fn * 16 + l16;
#pragma unroll
            for (int r = 0; r < 8; ++r) {
                const int m = bm0 + wm * 64 + fm * 16 + lhalf * 8 + r;
                out[(size_t)m * OUT_F + o] = acc[fm][fn][r];
            }
        }
    }
}

extern "C" void kernel_launch(void* const* d_in, const int* in_sizes, int n_in,
                              void* d_out, int out_size, void* d_ws, size_t ws_size,
                              hipStream_t stream) {
    const float* x      = (const float*)d_in[0];   // [4096, 4096] fp32
    const int*   qw     = (const int*)  d_in[1];   // [11008, 4096] int codes 0..15
    const float* scales = (const float*)d_in[2];   // [11008, 32] fp32
    float*       out    = (float*)d_out;           // [4096, 11008] fp32

    dim3 grid(OUT_F / BO, M_ROWS / BM);            // 43 x 16 blocks
    dim3 block(512);                               // 16 waves (wave32)
    rtn_marlin_wmma_bf16<<<grid, block, 0, stream>>>(x, qw, scales, out);
}
